// ConvFilter_44744969290299
// MI455X (gfx1250) — matrix operand order repaired and checked by reading, not compiled, not run on hardware
//
#include <hip/hip_runtime.h>

// Problem constants (from reference)
#define B_   64
#define S_   1024
#define F_   256
#define L_   1017          // S - K + 1
#define KTOT 2048          // im2col reduction length = F*K

// Tiling
#define BM 128             // rows (b,l) per workgroup (8 waves x 16)
#define BN 64              // out-channels per workgroup
#define KT 32              // K step per WMMA
#define NSTEP (KTOT / KT)  // 64
#define W_STRIDE 36        // shorts per LDS W row (32 data + 4 pad = 72B, matches TDM pad)
#define WBUF_SHORTS (BN * W_STRIDE)

// ---------------------------------------------------------------------------
// Rearrange conv_w[o][i][k] (fp32) -> Wt_t[o][r = k*256 + i] bf16 hi/lo split
// (o-major so a [n][k] tile is a plain 2D tile for the TDM)
// ---------------------------------------------------------------------------
__global__ __launch_bounds__(256) void prep_weights_kernel(
    const float* __restrict__ w,
    unsigned short* __restrict__ wt_hi,
    unsigned short* __restrict__ wt_lo) {
#if defined(__HIP_DEVICE_COMPILE__)
  int idx = blockIdx.x * 256 + threadIdx.x;   // idx = o*2048 + r
  int o = idx >> 11;
  int r = idx & 2047;
  int i = r & 255;
  int k = r >> 8;
  float v = w[(o * 256 + i) * 8 + k];
  __bf16 h  = (__bf16)v;
  __bf16 lo = (__bf16)(v - (float)h);
  unsigned short hu, lu;
  __builtin_memcpy(&hu, &h, 2);
  __builtin_memcpy(&lu, &lo, 2);
  wt_hi[idx] = hu;
  wt_lo[idx] = lu;
#endif
}

// ---------------------------------------------------------------------------
// Pre-split x (fp32) into bf16 hi/lo planes (same [B][S][F] layout)
// ---------------------------------------------------------------------------
__global__ __launch_bounds__(256) void prep_x_kernel(
    const float* __restrict__ x,
    unsigned short* __restrict__ xh,
    unsigned short* __restrict__ xl) {
#if defined(__HIP_DEVICE_COMPILE__)
  size_t idx = (size_t)blockIdx.x * 256 + threadIdx.x;
  float v = x[idx];
  __bf16 h  = (__bf16)v;
  __bf16 lo = (__bf16)(v - (float)h);
  unsigned short hu, lu;
  __builtin_memcpy(&hu, &h, 2);
  __builtin_memcpy(&lu, &lo, 2);
  xh[idx] = hu;
  xl[idx] = lu;
#endif
}

#if defined(__HIP_DEVICE_COMPILE__)

#define HAS_TDM __has_builtin(__builtin_amdgcn_tensor_load_to_lds)

typedef __attribute__((ext_vector_type(4))) unsigned int u32x4;
typedef __attribute__((ext_vector_type(8))) int          i32x8;
typedef __attribute__((ext_vector_type(4))) int          i32x4;
typedef __attribute__((ext_vector_type(16))) __bf16      v16bf;
typedef __attribute__((ext_vector_type(8)))  float       v8f;
typedef __attribute__((ext_vector_type(4)))  float       v4f;

union Frag { v16bf v; unsigned int u[8]; u32x4 q[2]; };

__device__ __forceinline__ unsigned pack_bf16(__bf16 a, __bf16 b) {
  unsigned short ua, ub;
  __builtin_memcpy(&ua, &a, 2);
  __builtin_memcpy(&ub, &b, 2);
  return (unsigned)ua | ((unsigned)ub << 16);
}

#if HAS_TDM
// TDM: load a 64(o) x 32(k) bf16 tile (row stride 2048 elems) into LDS with
// padding: 16 dwords data + 2 dwords pad -> 72B LDS row stride.
__device__ __forceinline__ void tdm_load_tile(const unsigned short* gptr,
                                              unsigned lds_off) {
  unsigned long long ga = (unsigned long long)(uintptr_t)gptr;
  u32x4 g0;
  g0.x = 1u;                                   // count=1 (valid user descriptor)
  g0.y = lds_off;                              // lds_addr (bytes)
  g0.z = (unsigned)(ga & 0xFFFFFFFFu);         // global_addr[31:0]
  g0.w = (unsigned)((ga >> 32) & 0x1FFFFFFu)   // global_addr[56:32]
         | (2u << 30);                         // type = 2 ("image")
  i32x8 g1;
  g1[0] = (1 << 16)        // data_size = 1 (2 bytes)
        | (1 << 20)        // pad_enable
        | (3 << 22)        // pad_interval code 3 = 16 dwords (64B row)
        | (1 << 25);       // pad_amount  code 1 = 2 dwords (8B)
  g1[1] = (int)((2048u & 0xFFFFu) << 16);  // tensor_dim0[15:0] = 2048
  g1[2] = (int)((256u & 0xFFFFu) << 16);   // dim0 hi=0 | tensor_dim1[15:0] = 256
  g1[3] = (int)(32u << 16);                // dim1 hi=0 | tile_dim0 = 32
  g1[4] = 64;                              // tile_dim1 = 64, tile_dim2 = 0
  g1[5] = 2048;                            // tensor_dim0_stride[31:0] = 2048 elems
  g1[6] = 0;
  g1[7] = 0;
  i32x4 z4 = {0, 0, 0, 0};
#if __clang_major__ >= 23
  i32x8 z8 = {0, 0, 0, 0, 0, 0, 0, 0};
  __builtin_amdgcn_tensor_load_to_lds(g0, g1, z4, z4, z8, 0);
#else
  __builtin_amdgcn_tensor_load_to_lds(g0, g1, z4, z4, 0);
#endif
}
#endif // HAS_TDM

#endif // __HIP_DEVICE_COMPILE__

// ---------------------------------------------------------------------------
// Fused: im2col GEMM (bf16 hi/lo split WMMA; A from global — either
// pre-split bf16 planes (PRESPLIT) or fp32 with in-register split;
// W double-buffered into LDS via TDM) + sigmoid + geometric filter.
// grid: (F/BN, ceil(L/BM), B), block: 256 (8 waves; wave w owns rows w*16..)
// ---------------------------------------------------------------------------
template <bool PRESPLIT>
__global__ __launch_bounds__(256) void conv_filter_kernel(
    const float* __restrict__ x,
    const unsigned short* __restrict__ xh,      // bf16 hi plane (PRESPLIT only)
    const unsigned short* __restrict__ xl,      // bf16 lo plane (PRESPLIT only)
    const unsigned short* __restrict__ wt_hi,   // [256][2048] bf16 (o-major)
    const unsigned short* __restrict__ wt_lo,
    const float* __restrict__ bias,
    float* __restrict__ out) {
#if defined(__HIP_DEVICE_COMPILE__)
  __shared__ __align__(16) unsigned short lds_w[2][2][WBUF_SHORTS]; // [buf][hi/lo]

  const int tid   = threadIdx.x;
  const int wave  = tid >> 5;
  const int lane  = tid & 31;
  const int lm    = lane & 15;
  const int lg    = lane >> 4;
  const int b     = blockIdx.z;
  const int l0    = blockIdx.y * BM;
  const int nbase = blockIdx.x * BN;

  const float* xb = x + ((size_t)b << 18);      // b * S * F
  int lrow = l0 + wave * 16 + lm;               // this lane's im2col row
  if (lrow > L_ - 1) lrow = L_ - 1;             // clamp (predicated out later)
  const size_t rowoff = (((size_t)b << 10) + lrow) * 256;  // contiguous window

  const size_t wbase = (size_t)nbase * KTOT;    // tile row start in Wt_t

  v8f c[4] = {};

  // ---- prologue: stage W tile 0 into buffer 0 ----
#if HAS_TDM
  if (wave == 0) {
    tdm_load_tile(wt_hi + wbase, (unsigned)(uintptr_t)&lds_w[0][0][0]);
    tdm_load_tile(wt_lo + wbase, (unsigned)(uintptr_t)&lds_w[0][1][0]);
    __builtin_amdgcn_s_wait_tensorcnt((short)0);
  }
#else
  for (int j = 0; j < 8; ++j) {                 // fallback: cooperative copy
    int e = j * 256 + tid;
    int nn = e >> 5, kk = e & 31;
    lds_w[0][0][nn * W_STRIDE + kk] = wt_hi[wbase + (size_t)nn * KTOT + kk];
    lds_w[0][1][nn * W_STRIDE + kk] = wt_lo[wbase + (size_t)nn * KTOT + kk];
  }
#endif
  __syncthreads();

  for (int kt = 0; kt < NSTEP; ++kt) {
    const int cur = kt & 1;
    const int nxt = cur ^ 1;
    const int k0  = kt * KT;

    // ---- kick off DMA of the NEXT W tile while we compute on the current ----
    if (kt + 1 < NSTEP) {
#if HAS_TDM
      if (wave == 0) {
        tdm_load_tile(wt_hi + wbase + k0 + KT,
                      (unsigned)(uintptr_t)&lds_w[nxt][0][0]);
        tdm_load_tile(wt_lo + wbase + k0 + KT,
                      (unsigned)(uintptr_t)&lds_w[nxt][1][0]);
      }
#else
      for (int j = 0; j < 8; ++j) {
        int e = j * 256 + tid;
        int nn = e >> 5, kk = e & 31;
        lds_w[nxt][0][nn * W_STRIDE + kk] =
            wt_hi[wbase + (size_t)nn * KTOT + k0 + KT + kk];
        lds_w[nxt][1][nn * W_STRIDE + kk] =
            wt_lo[wbase + (size_t)nn * KTOT + k0 + KT + kk];
      }
#endif
    }

    // ---- A fragments straight from global (contiguous im2col row) ----
    // lane holds row m=lane%16; k values lg*8+{0..7} and 16+lg*8+{0..7}
    Frag a_hi, a_lo;
    if constexpr (PRESPLIT) {
      const unsigned short* ph = xh + rowoff + k0 + lg * 8;
      const unsigned short* pl = xl + rowoff + k0 + lg * 8;
      a_hi.q[0] = *(const u32x4*)(ph);          // 8 bf16 = one b128
      a_hi.q[1] = *(const u32x4*)(ph + 16);
      a_lo.q[0] = *(const u32x4*)(pl);
      a_lo.q[1] = *(const u32x4*)(pl + 16);
      if (kt + 1 < NSTEP) {
        __builtin_prefetch(ph + KT, 0, 1);      // global_prefetch_b8
        __builtin_prefetch(pl + KT, 0, 1);
      }
    } else {
      const float* pa = x + rowoff + k0 + lg * 8;
      v4f f0 = *(const v4f*)(pa);
      v4f f1 = *(const v4f*)(pa + 4);
      v4f f2 = *(const v4f*)(pa + 16);
      v4f f3 = *(const v4f*)(pa + 20);
      float f[16] = {f0.x, f0.y, f0.z, f0.w, f1.x, f1.y, f1.z, f1.w,
                     f2.x, f2.y, f2.z, f2.w, f3.x, f3.y, f3.z, f3.w};
#pragma unroll
      for (int j = 0; j < 8; ++j) {
        float e0 = f[2 * j], e1 = f[2 * j + 1];
        __bf16 h0 = (__bf16)e0, h1 = (__bf16)e1;
        __bf16 q0 = (__bf16)(e0 - (float)h0), q1 = (__bf16)(e1 - (float)h1);
        a_hi.u[j] = pack_bf16(h0, h1);
        a_lo.u[j] = pack_bf16(q0, q1);
      }
      if (kt + 1 < NSTEP) __builtin_prefetch(pa + KT, 0, 1);
    }

    // ---- 4 N-subtiles, 3 split WMMAs each (hi*hi + hi*lo + lo*hi) ----
    const unsigned short* wh = lds_w[cur][0];
    const unsigned short* wl = lds_w[cur][1];
#pragma unroll
    for (int nt = 0; nt < 4; ++nt) {
      Frag b_hi, b_lo;
      const int wrow = (nt * 16 + lm) * W_STRIDE + (lg << 3);
#pragma unroll
      for (int j2 = 0; j2 < 8; ++j2) {
        b_hi.u[j2] = *(const unsigned int*)&wh[wrow + 2 * j2 + (j2 >= 4 ? 8 : 0)];
        b_lo.u[j2] = *(const unsigned int*)&wl[wrow + 2 * j2 + (j2 >= 4 ? 8 : 0)];
      }
      c[nt] = __builtin_amdgcn_wmma_f32_16x16x32_bf16(false, a_hi.v, false, b_hi.v,
                                                      (short)0, c[nt], false, false);
      c[nt] = __builtin_amdgcn_wmma_f32_16x16x32_bf16(false, a_hi.v, false, b_lo.v,
                                                      (short)0, c[nt], false, false);
      c[nt] = __builtin_amdgcn_wmma_f32_16x16x32_bf16(false, a_lo.v, false, b_hi.v,
                                                      (short)0, c[nt], false, false);
    }

    // ---- publish next W buffer ----
#if HAS_TDM
    if (wave == 0) __builtin_amdgcn_s_wait_tensorcnt((short)0);
#endif
    __syncthreads();
  }

  // ---- fused epilogue: sigmoid + geometric filter + window sum ----
  // C layout: VGPR j -> M = lg*8 + j, N = lm
#pragma unroll
  for (int nt = 0; nt < 4; ++nt) {
    const int f  = nbase + nt * 16 + lm;
    const float cb = bias[f];
#pragma unroll
    for (int j = 0; j < 8; ++j) {
      const int l = l0 + wave * 16 + lg * 8 + j;
      if (l < L_) {
        float t   = c[nt][j] + cb;
        float tau = 1.0f / (1.0f + __expf(-t));
        const float* xr = xb + (size_t)l * 256 + f;
        float p = 1.0f, den = 0.0f, num = 0.0f;
#pragma unroll
        for (int i = 7; i >= 0; --i) {           // w_i = tau^(7-i)
          num = fmaf(p, xr[i * 256], num);
          den += p;
          p *= tau;
        }
        out[((size_t)b * L_ + l) * 256 + f] = num / den;
      }
    }
  }
#endif // __HIP_DEVICE_COMPILE__
}

// ---------------------------------------------------------------------------
extern "C" void kernel_launch(void* const* d_in, const int* in_sizes, int n_in,
                              void* d_out, int out_size, void* d_ws, size_t ws_size,
                              hipStream_t stream) {
  const float* x      = (const float*)d_in[0];
  const float* conv_w = (const float*)d_in[1];
  const float* conv_b = (const float*)d_in[2];
  float* out = (float*)d_out;

  const size_t w_elems = (size_t)KTOT * F_;          // 524288 per plane
  const size_t x_elems = (size_t)B_ * S_ * F_;       // 16777216 per plane

  unsigned short* wt_hi = (unsigned short*)d_ws;     // [256][2048] bf16
  unsigned short* wt_lo = wt_hi + w_elems;
  unsigned short* xh    = wt_lo + w_elems;           // bf16 hi plane of x
  unsigned short* xl    = xh + x_elems;              // bf16 lo plane of x

  const size_t need = (2 * w_elems + 2 * x_elems) * sizeof(unsigned short);
  const bool presplit = ws_size >= need;

  prep_weights_kernel<<<(KTOT * F_) / 256, 256, 0, stream>>>(conv_w, wt_hi, wt_lo);

  dim3 grid(F_ / BN, (L_ + BM - 1) / BM, B_);
  if (presplit) {
    prep_x_kernel<<<(unsigned)(x_elems / 256), 256, 0, stream>>>(x, xh, xl);
    conv_filter_kernel<true><<<grid, dim3(256), 0, stream>>>(
        x, xh, xl, wt_hi, wt_lo, conv_b, out);
  } else {
    conv_filter_kernel<false><<<grid, dim3(256), 0, stream>>>(
        x, nullptr, nullptr, wt_hi, wt_lo, conv_b, out);
  }
}
